// FastRNNDetHead_88802743812869
// MI455X (gfx1250) — matrix operand order repaired and checked by reading, not compiled, not run on hardware
//
#include <hip/hip_runtime.h>
#include <hip/hip_bf16.h>
#include <stdint.h>

// ---------------------------------------------------------------------------
// Types for CDNA5 WMMA (wave32): D = A(16x32 bf16) * B(32x16 bf16) + C(f32)
// ---------------------------------------------------------------------------
typedef __attribute__((ext_vector_type(16))) __bf16       bf16x16;
typedef __attribute__((ext_vector_type(8)))  float        f32x8;
typedef __attribute__((ext_vector_type(8)))  unsigned int u32x8;

__device__ __forceinline__ unsigned short f2bf(float f) {
    unsigned int u = __builtin_bit_cast(unsigned int, f);
    unsigned int r = u + 0x7FFFu + ((u >> 16) & 1u);   // round-to-nearest-even
    return (unsigned short)(r >> 16);
}

__device__ __forceinline__ unsigned int pack2bf(float lo, float hi) {
    return (unsigned int)f2bf(lo) | ((unsigned int)f2bf(hi) << 16);
}

__device__ __forceinline__ f32x8 zero8() {
    f32x8 z = {0.f, 0.f, 0.f, 0.f, 0.f, 0.f, 0.f, 0.f};
    return z;
}

// A fragment (16x32, MxK) from row-major bf16 source, leading dim lda elems.
// Layout: lanes 0-15 -> M=lane, V0..7=K0..15;
//         lanes 16-31 -> M=lane-16, V0..7=K16..31.
__device__ __forceinline__ bf16x16 load_a(const unsigned short* A, int lda,
                                          int m_base, int k0, int lane) {
    int row  = m_base + (lane & 15);
    int koff = (lane & 16) ? 16 : 0;
    const unsigned int* q = (const unsigned int*)(A + (size_t)row * lda + k0 + koff);
    u32x8 r;
    r[0] = q[0]; r[1] = q[1]; r[2] = q[2];  r[3] = q[3];
    r[4] = q[4]; r[5] = q[5]; r[6] = q[6];  r[7] = q[7];
    return __builtin_bit_cast(bf16x16, r);
}

// B fragment (32x16, KxN) from weights stored (N,K) row-major (i.e. "@ W.T").
// Layout (contiguous k, same as load_a): lanes 0-15 -> N=lane holding K0..15 packed ascending in V0..7;
//             lanes 16-31 -> N=lane-16 holding K16..31.
__device__ __forceinline__ bf16x16 load_b(const unsigned short* Bw, int ldb,
                                          int n_base, int k0, int lane) {
    int col  = n_base + (lane & 15);
    int krow = (lane & 16) ? 16 : 0;
    const unsigned int* q = (const unsigned int*)(Bw + (size_t)col * ldb + k0 + krow);
    u32x8 r;
#pragma unroll
    for (int i = 0; i < 8; ++i) r[i] = q[i];
    return __builtin_bit_cast(bf16x16, r);
}

__device__ __forceinline__ bf16x16 load_b_guard(const unsigned short* Bw, int ldb,
                                                int n_base, int k0, int lane,
                                                int n_valid) {
    int col  = n_base + (lane & 15);
    int krow = (lane & 16) ? 16 : 0;
    u32x8 r;
    if (col < n_valid) {
        const unsigned int* q = (const unsigned int*)(Bw + (size_t)col * ldb + k0 + krow);
#pragma unroll
        for (int i = 0; i < 8; ++i) r[i] = q[i];
    } else {
#pragma unroll
        for (int i = 0; i < 8; ++i) r[i] = 0u;
    }
    return __builtin_bit_cast(bf16x16, r);
}

// B fragment built directly from an f32 (N,K) row-major matrix (convert+pack).
__device__ __forceinline__ bf16x16 make_b_from_f32(const float* Bw, int ldb,
                                                   int n_base, int k0, int lane) {
    int col  = n_base + (lane & 15);
    int krow = (lane & 16) ? 16 : 0;
    const float* p = Bw + (size_t)col * ldb + k0 + krow;
    u32x8 r;
#pragma unroll
    for (int i = 0; i < 8; ++i) r[i] = pack2bf(p[2 * i], p[2 * i + 1]);
    return __builtin_bit_cast(bf16x16, r);
}

// ---------------------------------------------------------------------------
// Elementwise f32 -> bf16 conversion (weights)
// ---------------------------------------------------------------------------
__global__ void cvt_f32_bf16_kernel(const float* __restrict__ in,
                                    unsigned short* __restrict__ out, int n) {
    int i = blockIdx.x * blockDim.x + threadIdx.x;
    if (i < n) out[i] = f2bf(in[i]);
}

// ---------------------------------------------------------------------------
// feat (B=16, C=256, S=4096) f32 -> token-major (B*S, C) bf16 via LDS tiles
// ---------------------------------------------------------------------------
__global__ void transpose_feat_kernel(const float* __restrict__ feat,
                                      unsigned short* __restrict__ out) {
    __shared__ float tile[32][33];
    const int b  = blockIdx.z;
    const int c0 = blockIdx.y * 32;
    const int s0 = blockIdx.x * 32;
    const int tx = threadIdx.x;      // 0..31
    const int ty = threadIdx.y;      // 0..7
#pragma unroll
    for (int i = ty; i < 32; i += 8)
        tile[i][tx] = feat[((size_t)(b * 256 + c0 + i)) * 4096 + s0 + tx];
    __syncthreads();
#pragma unroll
    for (int i = ty; i < 32; i += 8) {
        int s = s0 + i, c = c0 + tx;
        out[((size_t)(b * 4096 + s)) * 256 + c] = f2bf(tile[tx][i]);
    }
}

// ---------------------------------------------------------------------------
// Generic WMMA GEMM: C[M,N] = A[M,K](bf16) * Bw[N,K]^T(bf16) + bias
// Each wave computes a 16x64 strip (A-frag reused across 4 N-tiles).
// Ping/pong software pipeline over 64-wide k-chunks: loads write directly
// into the alternate fragment set (no register copies, no in-loop branch).
// Requires K % 64 == 0 (K = 256 / 512 here). The final P-reload is clamped
// to k=0: a redundant, unused, cache-hit load instead of a branch.
// mode: 0 = f32 out, 1 = bf16 out, 2 = bf16 out + ReLU
// ---------------------------------------------------------------------------
__global__ void __launch_bounds__(256)
wmma_gemm_kernel(const unsigned short* __restrict__ A,
                 const unsigned short* __restrict__ Bw,
                 const float* __restrict__ bias,
                 void* __restrict__ Cout,
                 int M, int N, int K, int mode) {
    const int gwave  = (blockIdx.x * blockDim.x + threadIdx.x) >> 5;
    const int lane   = threadIdx.x & 31;
    const int mtiles = M >> 4;
    const int strip  = gwave / mtiles;
    const int mt     = gwave % mtiles;
    if (strip >= (N >> 6)) return;
    const int m_base = mt << 4;
    const int n_base = strip << 6;

    f32x8 acc[4];
#pragma unroll
    for (int t = 0; t < 4; ++t) acc[t] = zero8();

    // prologue: P fragments for k = 0
    bf16x16 aP = load_a(A, K, m_base, 0, lane);
    bf16x16 bP[4];
#pragma unroll
    for (int t = 0; t < 4; ++t) bP[t] = load_b(Bw, K, n_base + t * 16, 0, lane);

    for (int k0 = 0; k0 < K; k0 += 64) {
        const int k1 = k0 + 32;
        bf16x16 aQ = load_a(A, K, m_base, k1, lane);
        bf16x16 bQ[4];
#pragma unroll
        for (int t = 0; t < 4; ++t) bQ[t] = load_b(Bw, K, n_base + t * 16, k1, lane);
#pragma unroll
        for (int t = 0; t < 4; ++t)
            acc[t] = __builtin_amdgcn_wmma_f32_16x16x32_bf16(
                         false, aP, false, bP[t], (short)0, acc[t], false, false);

        const int k2 = (k0 + 64 < K) ? (k0 + 64) : 0;   // clamped (last is unused)
        aP = load_a(A, K, m_base, k2, lane);
#pragma unroll
        for (int t = 0; t < 4; ++t) bP[t] = load_b(Bw, K, n_base + t * 16, k2, lane);
#pragma unroll
        for (int t = 0; t < 4; ++t)
            acc[t] = __builtin_amdgcn_wmma_f32_16x16x32_bf16(
                         false, aQ, false, bQ[t], (short)0, acc[t], false, false);
    }

    const int l15 = lane & 15;
    const int mh  = (lane & 16) ? 8 : 0;
#pragma unroll
    for (int t = 0; t < 4; ++t) {
        int   n  = n_base + t * 16 + l15;
        float bv = bias[n];
#pragma unroll
        for (int r = 0; r < 8; ++r) {
            int   m = m_base + r + mh;
            float v = acc[t][r] + bv;
            if (mode == 0) {
                ((float*)Cout)[(size_t)m * N + n] = v;
            } else {
                if (mode == 2) v = fmaxf(v, 0.f);
                ((unsigned short*)Cout)[(size_t)m * N + n] = f2bf(v);
            }
        }
    }
}

// ---------------------------------------------------------------------------
// Head second layer: out[b, n, s] = A[t=b*4096+s, 0:512] . w2[n, 0:512] + b2
// Writes directly in (B, N_out, H, W) layout. Optional ReLU (reg head).
// Same branch-free ping/pong pipeline (K = 512).
// ---------------------------------------------------------------------------
__global__ void __launch_bounds__(256)
wmma_head2_kernel(const unsigned short* __restrict__ A,
                  const unsigned short* __restrict__ Bw,
                  const float* __restrict__ bias,
                  float* __restrict__ out, int N_out, int relu) {
    const int K      = 512;
    const int gwave  = (blockIdx.x * blockDim.x + threadIdx.x) >> 5;
    const int lane   = threadIdx.x & 31;
    const int mtiles = 65536 >> 4;
    const int nt     = gwave / mtiles;
    const int mt     = gwave % mtiles;
    if (nt >= ((N_out + 15) >> 4)) return;
    const int m_base = mt << 4;
    const int n_base = nt << 4;

    f32x8 acc = zero8();
    bf16x16 aP = load_a(A, K, m_base, 0, lane);
    bf16x16 bP = load_b_guard(Bw, K, n_base, 0, lane, N_out);
    for (int k0 = 0; k0 < K; k0 += 64) {
        const int k1 = k0 + 32;
        bf16x16 aQ = load_a(A, K, m_base, k1, lane);
        bf16x16 bQ = load_b_guard(Bw, K, n_base, k1, lane, N_out);
        acc = __builtin_amdgcn_wmma_f32_16x16x32_bf16(
                  false, aP, false, bP, (short)0, acc, false, false);

        const int k2 = (k0 + 64 < K) ? (k0 + 64) : 0;   // clamped (last is unused)
        aP = load_a(A, K, m_base, k2, lane);
        bP = load_b_guard(Bw, K, n_base, k2, lane, N_out);
        acc = __builtin_amdgcn_wmma_f32_16x16x32_bf16(
                  false, aQ, false, bQ, (short)0, acc, false, false);
    }

    const int n = n_base + (lane & 15);
    if (n >= N_out) return;
    const float bv = bias[n];
    const int   mh = (lane & 16) ? 8 : 0;
#pragma unroll
    for (int r = 0; r < 8; ++r) {
        int   t = m_base + r + mh;
        int   b = t >> 12;        // / 4096
        int   s = t & 4095;
        float v = acc[r] + bv;
        if (relu) v = fmaxf(v, 0.f);
        out[((size_t)b * N_out + n) * 4096 + s] = v;
    }
}

// ---------------------------------------------------------------------------
// Bidirectional FastRNN scan. grid.x = 2 (dir 0 = forward, 1 = backward).
// Block = 512 threads = 16 waves; wave w owns N-tile [16w, 16w+16).
// U's B-fragments are loop-invariant: converted f32->bf16 once into 64 VGPRs
// per lane and kept register-resident for all 4096 steps. LDS holds only h
// (bf16 for WMMA A-frags + f32 for the beta*h term) = 24KB static.
// Per step: issue all 8 A-fragment DS loads, then 8 back-to-back WMMAs so
// DS waits stagger across the matrix-op chain.
// h_new = sig(alpha)*tanh(pre[s] + h @ U^T) + sig(beta)*h
// Backward dir iterates s = 4095..0 (equivalent to scanning reversed pre and
// un-reversing the outputs). Writes bf16 h into h_cat[:, col_off+n].
// ---------------------------------------------------------------------------
__global__ void __launch_bounds__(512)
fastrnn_scan_kernel(const float* __restrict__ Uf, const float* __restrict__ preF,
                    const float* __restrict__ alphaF, const float* __restrict__ betaF,
                    const float* __restrict__ Ubk, const float* __restrict__ preB,
                    const float* __restrict__ alphaB, const float* __restrict__ betaB,
                    unsigned short* __restrict__ h_cat) {
    __shared__ unsigned short hb[16 * 256];   // h in bf16 (WMMA A source)
    __shared__ float          hf[16 * 256];   // h in f32  (beta*h term)

    const int    dir = blockIdx.x;
    const float* U   = dir ? Ubk : Uf;
    const float* pre = dir ? preB : preF;
    const float  a   = 1.f / (1.f + expf(-(dir ? alphaB[0] : alphaF[0])));
    const float  bC  = 1.f / (1.f + expf(-(dir ? betaB[0] : betaF[0])));
    const int    col_off = dir ? 256 : 0;

    const int tid    = threadIdx.x;
    const int wid    = tid >> 5;
    const int lane   = tid & 31;
    const int n_base = wid << 4;
    const int l15    = lane & 15;
    const int mh     = (lane & 16) ? 8 : 0;

    // Hoist U's B-fragments into registers (loop-invariant across all steps).
    bf16x16 bU[8];
#pragma unroll
    for (int kc = 0; kc < 8; ++kc)
        bU[kc] = make_b_from_f32(U, 256, n_base, kc * 32, lane);

    for (int i = tid; i < 16 * 256; i += 512) { hf[i] = 0.f; hb[i] = 0; }
    __syncthreads();

    for (int step = 0; step < 4096; ++step) {
        const int s = dir ? (4095 - step) : step;

        // issue all A-fragment LDS loads first, then the WMMA chain
        bf16x16 af[8];
#pragma unroll
        for (int kc = 0; kc < 8; ++kc)
            af[kc] = load_a(hb, 256, 0, kc * 32, lane);

        f32x8 acc = zero8();
#pragma unroll
        for (int kc = 0; kc < 8; ++kc)
            acc = __builtin_amdgcn_wmma_f32_16x16x32_bf16(
                      false, af[kc], false, bU[kc], (short)0, acc, false, false);

        __syncthreads();   // all waves done reading hb before it is rewritten
#pragma unroll
        for (int r = 0; r < 8; ++r) {
            int   m    = r + mh;                 // batch row
            int   n    = n_base + l15;           // hidden column
            float pv   = pre[((size_t)(m * 4096 + s)) * 256 + n];
            float hold = hf[m * 256 + n];
            float hn   = a * tanhf(pv + acc[r]) + bC * hold;
            hf[m * 256 + n] = hn;
            unsigned short hv = f2bf(hn);
            hb[m * 256 + n] = hv;
            h_cat[((size_t)(m * 4096 + s)) * 512 + col_off + n] = hv;
        }
        __syncthreads();
    }
}

// ---------------------------------------------------------------------------
// Host launcher
// ---------------------------------------------------------------------------
extern "C" void kernel_launch(void* const* d_in, const int* in_sizes, int n_in,
                              void* d_out, int out_size, void* d_ws, size_t ws_size,
                              hipStream_t stream) {
    (void)in_sizes; (void)n_in; (void)out_size; (void)ws_size;

    const float* feat    = (const float*)d_in[0];
    const float* proj_w  = (const float*)d_in[1];
    const float* proj_b  = (const float*)d_in[2];
    const float* Wf      = (const float*)d_in[3];
    const float* Uf      = (const float*)d_in[4];
    const float* bf      = (const float*)d_in[5];
    const float* alpha_f = (const float*)d_in[6];
    const float* beta_f  = (const float*)d_in[7];
    const float* Wb      = (const float*)d_in[8];
    const float* Ub      = (const float*)d_in[9];
    const float* bb      = (const float*)d_in[10];
    const float* alpha_b = (const float*)d_in[11];
    const float* beta_b  = (const float*)d_in[12];
    const float* cls_w1  = (const float*)d_in[13];
    const float* cls_b1  = (const float*)d_in[14];
    const float* cls_w2  = (const float*)d_in[15];
    const float* cls_b2  = (const float*)d_in[16];
    const float* reg_w1  = (const float*)d_in[17];
    const float* reg_b1  = (const float*)d_in[18];
    const float* reg_w2  = (const float*)d_in[19];
    const float* reg_b2  = (const float*)d_in[20];
    const float* ctr_w1  = (const float*)d_in[21];
    const float* ctr_b1  = (const float*)d_in[22];
    const float* ctr_w2  = (const float*)d_in[23];
    const float* ctr_b2  = (const float*)d_in[24];

    char* ws = (char*)d_ws;
    const size_t MB = 1ull << 20;
    unsigned short* A_t   = (unsigned short*)(ws);             // 32MB (dead after proj)
    unsigned short* x_bf  = (unsigned short*)(ws + 32 * MB);   // 32MB (dead after pre)
    float*          preF  = (float*)(ws + 64 * MB);            // 64MB
    float*          preB  = (float*)(ws + 128 * MB);           // 64MB
    unsigned short* h_cat = (unsigned short*)(ws + 192 * MB);  // 64MB
    unsigned short* t1    = (unsigned short*)(ws);             // 64MB alias over A_t+x

    char* wr = ws + 256 * MB;                                  // bf16 weights (~2.2MB)
    unsigned short* pw  = (unsigned short*)wr;   // 256*256
    unsigned short* wfb = pw  + 65536;           // 256*256
    unsigned short* wbb = wfb + 65536;           // 256*256
    unsigned short* c1  = wbb + 65536;           // 512*512
    unsigned short* r1  = c1  + 262144;
    unsigned short* g1  = r1  + 262144;          // ctr_w1
    unsigned short* c2  = g1  + 262144;          // 80*512
    unsigned short* r2  = c2  + 40960;           // 4*512
    unsigned short* g2  = r2  + 2048;            // 1*512

    auto cvt = [&](const float* src, unsigned short* dst, int n) {
        cvt_f32_bf16_kernel<<<(n + 255) / 256, 256, 0, stream>>>(src, dst, n);
    };
    cvt(proj_w, pw, 65536);
    cvt(Wf, wfb, 65536);
    cvt(Wb, wbb, 65536);
    cvt(cls_w1, c1, 262144);
    cvt(reg_w1, r1, 262144);
    cvt(ctr_w1, g1, 262144);
    cvt(cls_w2, c2, 40960);
    cvt(reg_w2, r2, 2048);
    cvt(ctr_w2, g2, 512);

    // feat (B,C,H,W) -> token-major bf16 (B*S, C)
    {
        dim3 tb(32, 8);
        dim3 tg(4096 / 32, 256 / 32, 16);
        transpose_feat_kernel<<<tg, tb, 0, stream>>>(feat, A_t);
    }

    // proj: x = feat_t @ proj_w^T + proj_b      (M=65536, N=256, K=256) -> bf16
    wmma_gemm_kernel<<<2048, 256, 0, stream>>>(A_t, pw, proj_b, (void*)x_bf,
                                               65536, 256, 256, 1);
    // pre_f = x @ Wf^T + bf  -> f32 ; pre_b = x @ Wb^T + bb -> f32
    wmma_gemm_kernel<<<2048, 256, 0, stream>>>(x_bf, wfb, bf, (void*)preF,
                                               65536, 256, 256, 0);
    wmma_gemm_kernel<<<2048, 256, 0, stream>>>(x_bf, wbb, bb, (void*)preB,
                                               65536, 256, 256, 0);

    // bidirectional FastRNN scan (both directions concurrent)
    fastrnn_scan_kernel<<<2, 512, 0, stream>>>(
        Uf, preF, alpha_f, beta_f, Ub, preB, alpha_b, beta_b, h_cat);

    float* out_cls = (float*)d_out;                       // 16*80*4096
    float* out_reg = out_cls + (size_t)16 * 80 * 4096;    // 16*4*4096
    float* out_ctr = out_reg + (size_t)16 * 4 * 4096;     // 16*1*4096

    // cls head
    wmma_gemm_kernel<<<4096, 256, 0, stream>>>(h_cat, c1, cls_b1, (void*)t1,
                                               65536, 512, 512, 2);
    wmma_head2_kernel<<<2560, 256, 0, stream>>>(t1, c2, cls_b2, out_cls, 80, 0);
    // reg head (ReLU on output)
    wmma_gemm_kernel<<<4096, 256, 0, stream>>>(h_cat, r1, reg_b1, (void*)t1,
                                               65536, 512, 512, 2);
    wmma_head2_kernel<<<512, 256, 0, stream>>>(t1, r2, reg_b2, out_reg, 4, 1);
    // ctr head
    wmma_gemm_kernel<<<4096, 256, 0, stream>>>(h_cat, g1, ctr_b1, (void*)t1,
                                               65536, 512, 512, 2);
    wmma_head2_kernel<<<512, 256, 0, stream>>>(t1, g2, ctr_b2, out_ctr, 1, 0);
}
